// Mamba_v3_12240656793952
// MI455X (gfx1250) — matrix operand order repaired and checked by reading, not compiled, not run on hardware
//
#include <hip/hip_runtime.h>
#include <hip/hip_bf16.h>

typedef __attribute__((ext_vector_type(16))) _Float16 v16h;
typedef __attribute__((ext_vector_type(8)))  _Float16 v8h;
typedef __attribute__((ext_vector_type(8)))  float    v8f;

#define LSEQ 16
#define DM   256
#define DI   512
#define BATCH 512

// ---------------------------------------------------------------------------
// WMMA helpers (gfx1250 wave32, 16x16x32 f16 -> f32)
// ---------------------------------------------------------------------------
__device__ __forceinline__ v8f wmma_f16(v16h a, v16h b, v8f c) {
  return __builtin_amdgcn_wmma_f32_16x16x32_f16(
      /*neg_a=*/false, a, /*neg_b=*/false, b,
      /*c_mod=*/(short)0, c, /*reuse_a=*/false, /*reuse_b=*/false);
}

// A-matrix fragment, 16x32 f16, documented CDNA5 layout:
// lane%16 = M row; lane/16 selects K-half; VGPR v holds K pair
//   k0 = (v&3)*2 + (lane/16)*8 + (v>>2)*16
// (pattern = two contiguous 8-f16 chunks -> lowers to 2x b128)
__device__ __forceinline__ v16h loadA(const _Float16* a, int ld, int rowBase,
                                      int kBase, int lane) {
  const int m    = rowBase + (lane & 15);
  const int half = lane >> 4;
  const _Float16* p = a + (size_t)m * ld + kBase;
  v16h r;
#pragma unroll
  for (int v = 0; v < 8; ++v) {
    const int k0 = ((v & 3) << 1) + (half << 3) + ((v >> 2) << 4);
    r[2 * v]     = p[k0];
    r[2 * v + 1] = p[k0 + 1];
  }
  return r;
}

// B-matrix fragment, 32x16 f16 from row-major weight W(N,K):
// lane%16 = N column; lanes 0-15 hold K=0..7,16..23, lanes 16-31 hold K=8..15,24..31
// -> per lane two 8-f16 chunks = two b128 loads.
__device__ __forceinline__ v16h loadB(const _Float16* w, int ldk, int n0,
                                      int kBase, int lane) {
  const _Float16* p =
      w + (size_t)(n0 + (lane & 15)) * ldk + kBase + ((lane >> 4) << 3);
  v8h lo = *(const v8h*)p;
  v8h hi = *(const v8h*)(p + 16);
  v16h r;
#pragma unroll
  for (int e = 0; e < 8; ++e) { r[e] = lo[e]; r[e + 8] = hi[e]; }
  return r;
}

// ---------------------------------------------------------------------------
// Prep: f32 -> f16 weight conversion
// ---------------------------------------------------------------------------
__global__ void cvt_f16(const float* __restrict__ src, _Float16* __restrict__ dst,
                        int n) {
  for (int i = blockIdx.x * blockDim.x + threadIdx.x; i < n;
       i += gridDim.x * blockDim.x)
    dst[i] = (_Float16)src[i];
}

// dt_proj_w (512,16) -> padded (512,32) with zeros in K=16..31
__global__ void cvt_dtpad(const float* __restrict__ src,
                          _Float16* __restrict__ dst) {
  int i = blockIdx.x * blockDim.x + threadIdx.x;
  if (i < 512 * 32) {
    int row = i >> 5, k = i & 31;
    dst[i] = (k < 16) ? (_Float16)src[row * 16 + k] : (_Float16)0.f;
  }
}

// ---------------------------------------------------------------------------
// mods = F_clip_s(512x512) @ fs_w^T(512x1536) + fs_b    (WMMA)
// ---------------------------------------------------------------------------
__global__ __launch_bounds__(256) void mods_gemm(
    const _Float16* __restrict__ fclip, const _Float16* __restrict__ fsw,
    const float* __restrict__ fsb, float* __restrict__ mods) {
  const int tid = threadIdx.x, lane = tid & 31, wave = tid >> 5;
  const int tile = blockIdx.x * 8 + wave;           // 3072 tiles total
  const int m0 = (tile / 96) * 16;
  const int n0 = (tile % 96) * 16;
  v8f acc = {};
#pragma unroll 4
  for (int ks = 0; ks < 16; ++ks) {
    v16h a = loadA(fclip, 512, m0, ks * 32, lane);
    v16h b = loadB(fsw, 512, n0, ks * 32, lane);
    acc = wmma_f16(a, b, acc);
  }
  const int n = n0 + (lane & 15);
  const int mh = (lane >> 4) * 8;
#pragma unroll
  for (int r = 0; r < 8; ++r)
    mods[(size_t)(m0 + mh + r) * 1536 + n] = acc[r] + fsb[n];
}

// ---------------------------------------------------------------------------
// LayerNorm(d=256) + modulate -> f16 staging. 8 waves x 2 rows, shfl reduce.
// ---------------------------------------------------------------------------
__device__ __forceinline__ void layernorm_mod(const float* __restrict__ x,
                                              _Float16* __restrict__ o,
                                              const float* __restrict__ shift,
                                              const float* __restrict__ scale,
                                              int wave, int lane) {
#pragma unroll
  for (int rr = 0; rr < 2; ++rr) {
    const int row = wave * 2 + rr;
    const float* xr = x + row * DM;
    float v[8], s = 0.f;
#pragma unroll
    for (int j = 0; j < 8; ++j) { v[j] = xr[lane * 8 + j]; s += v[j]; }
#pragma unroll
    for (int m = 16; m >= 1; m >>= 1) s += __shfl_xor(s, m, 32);
    const float mean = s * (1.f / 256.f);
    float q = 0.f;
#pragma unroll
    for (int j = 0; j < 8; ++j) { float t = v[j] - mean; q += t * t; }
#pragma unroll
    for (int m = 16; m >= 1; m >>= 1) q += __shfl_xor(q, m, 32);
    const float rstd = rsqrtf(q * (1.f / 256.f) + 1e-6f);
#pragma unroll
    for (int j = 0; j < 8; ++j) {
      const int d = lane * 8 + j;
      const float ln = (v[j] - mean) * rstd;
      o[row * DM + d] = (_Float16)(ln * (1.f + scale[d]) + shift[d]);
    }
  }
}

// ---------------------------------------------------------------------------
// Fused Mamba block: one workgroup (8 waves) per batch element.
// ---------------------------------------------------------------------------
__global__ __launch_bounds__(256) void fused_block(
    const float* __restrict__ F_content, const float* __restrict__ mods,
    const _Float16* __restrict__ w_inproj, const float* __restrict__ conv_w,
    const float* __restrict__ conv_b, const _Float16* __restrict__ w_xproj,
    const _Float16* __restrict__ w_dtpad, const float* __restrict__ dt_b,
    const float* __restrict__ A_log, const float* __restrict__ D_skip,
    const _Float16* __restrict__ w_outproj, const _Float16* __restrict__ w_fc1,
    const float* __restrict__ fc1_b, const _Float16* __restrict__ w_fc2,
    const float* __restrict__ fc2_b, float* __restrict__ out) {
  extern __shared__ char smem[];
  float*    sXres  = (float*)smem;                    // 16x256 f32 residual
  _Float16* sA     = (_Float16*)(sXres + LSEQ * DM);  // 16x512 f16 A staging
  float*    sXb    = (float*)(sA + LSEQ * DI);        // 16x512 f32 x branch
  float*    sZb    = sXb + LSEQ * DI;                 // 16x512 f32 z branch
  float*    sDelta = sZb + LSEQ * DI;                 // 16x512 f32 delta (->y in place)
  float*    sDbl   = sDelta + LSEQ * DI;              // 16x48  f32 dt|B|C
  float*    sY     = sDelta;                          // alias: scan overwrites in place
  float*    sAtt   = sZb;                             // alias: z consumed before attn out

  const int b = blockIdx.x;
  const int tid = threadIdx.x, lane = tid & 31, wave = tid >> 5;
  const float* mrow = mods + (size_t)b * 1536;

  // 1) pixel-shuffle gather: x[l, d] = F_content[b, c, hc*8+r1, wc*8+r2]
  for (int i = tid; i < LSEQ * DM; i += 256) {
    const int l = i >> 8, d = i & 255;
    const int c = d >> 6, r1 = (d >> 3) & 7, r2 = d & 7;
    const int hc = l >> 2, wc = l & 3;
    sXres[i] =
        F_content[(((size_t)b * 4 + c) * 32 + hc * 8 + r1) * 32 + wc * 8 + r2];
  }
  __syncthreads();

  // 2) LN + modulate (mamba branch) -> sA f16 [16][256]
  layernorm_mod(sXres, sA, mrow + 0, mrow + 256, wave, lane);
  __syncthreads();

  // 3) in_proj: (16x256)@(256x1024) -> x|z. A fragments hoisted to registers,
  //    reused across all 8 N-tiles per wave.
  {
    v16h aF[8];
#pragma unroll
    for (int ks = 0; ks < 8; ++ks) aF[ks] = loadA(sA, DM, 0, ks * 32, lane);
    for (int t = wave; t < 64; t += 8) {
      const int n0 = t * 16;
      if (t + 8 < 64)  // hide L2 latency on the next weight tile
        __builtin_prefetch(w_inproj + (size_t)(t + 8) * 16 * DM, 0, 1);
      v8f acc = {};
#pragma unroll
      for (int ks = 0; ks < 8; ++ks)
        acc = wmma_f16(aF[ks], loadB(w_inproj, DM, n0, ks * 32, lane), acc);
      const int n = n0 + (lane & 15), mh = (lane >> 4) * 8;
#pragma unroll
      for (int r = 0; r < 8; ++r) {
        if (n < DI) sXb[(mh + r) * DI + n] = acc[r];
        else        sZb[(mh + r) * DI + (n - DI)] = acc[r];
      }
    }
  }
  __syncthreads();

  // 4) causal depthwise conv(k=4) + SiLU, per channel in registers
  for (int d = tid; d < DI; d += 256) {
    const float w0 = conv_w[d * 4 + 0], w1 = conv_w[d * 4 + 1];
    const float w2 = conv_w[d * 4 + 2], w3 = conv_w[d * 4 + 3];
    const float cb = conv_b[d];
    float xv[LSEQ];
#pragma unroll
    for (int l = 0; l < LSEQ; ++l) xv[l] = sXb[l * DI + d];
#pragma unroll
    for (int l = 0; l < LSEQ; ++l) {
      float s = cb + xv[l] * w3;
      if (l >= 1) s += xv[l - 1] * w2;
      if (l >= 2) s += xv[l - 2] * w1;
      if (l >= 3) s += xv[l - 3] * w0;
      sXb[l * DI + d] = s / (1.f + __expf(-s));
    }
  }
  __syncthreads();

  // 5) stage x (f16) for x_proj
  for (int i = tid; i < LSEQ * DI; i += 256) sA[i] = (_Float16)sXb[i];
  __syncthreads();

  // 6) x_proj: (16x512)@(512x48) -> dt|B|C. 3 tiles on waves 0..2.
  if (wave < 3) {
    const int n0 = wave * 16;
    v8f acc = {};
#pragma unroll 4
    for (int ks = 0; ks < 16; ++ks) {
      v16h a = loadA(sA, DI, 0, ks * 32, lane);
      v16h bm = loadB(w_xproj, DI, n0, ks * 32, lane);
      acc = wmma_f16(a, bm, acc);
    }
    const int n = n0 + (lane & 15), mh = (lane >> 4) * 8;
#pragma unroll
    for (int r = 0; r < 8; ++r) sDbl[(mh + r) * 48 + n] = acc[r];
  }
  __syncthreads();

  // 7) dt_proj (K=16 zero-padded to 32) + softplus -> sDelta
  for (int i = tid; i < 16 * 32; i += 256) {
    const int l = i >> 5, k = i & 31;
    sA[i] = (k < 16) ? (_Float16)sDbl[l * 48 + k] : (_Float16)0.f;
  }
  __syncthreads();
  {
    const v16h a = loadA(sA, 32, 0, 0, lane);  // shared by all 4 tiles
    for (int t = wave; t < 32; t += 8) {
      const int n0 = t * 16;
      v8f acc = {};
      acc = wmma_f16(a, loadB(w_dtpad, 32, n0, 0, lane), acc);
      const int n = n0 + (lane & 15), mh = (lane >> 4) * 8;
#pragma unroll
      for (int r = 0; r < 8; ++r) {
        const float pre = acc[r] + dt_b[n];
        sDelta[(mh + r) * DI + n] = (pre > 20.f) ? pre : log1pf(__expf(pre));
      }
    }
  }
  __syncthreads();

  // 8) selective scan: thread owns 2 channels; h-state and A in registers,
  //    l outer so the shared B/C rows are read from LDS once per step.
  {
    const int d0 = tid, d1 = tid + 256;
    float a0[16], a1[16], h0[16], h1[16];
    const float dsk0 = D_skip[d0], dsk1 = D_skip[d1];
#pragma unroll
    for (int n = 0; n < 16; ++n) {
      a0[n] = -__expf(A_log[d0 * 16 + n]);
      a1[n] = -__expf(A_log[d1 * 16 + n]);
      h0[n] = 0.f;
      h1[n] = 0.f;
    }
#pragma unroll
    for (int l = 0; l < LSEQ; ++l) {
      float bn[16], cn[16];
#pragma unroll
      for (int n = 0; n < 16; ++n) {
        bn[n] = sDbl[l * 48 + 16 + n];
        cn[n] = sDbl[l * 48 + 32 + n];
      }
      // channel d0
      {
        const float dl = sDelta[l * DI + d0];
        const float xv = sXb[l * DI + d0];
        const float dlx = dl * xv;
        float acc = 0.f;
#pragma unroll
        for (int n = 0; n < 16; ++n) {
          h0[n] = __expf(dl * a0[n]) * h0[n] + dlx * bn[n];
          acc += h0[n] * cn[n];
        }
        float y = acc + dsk0 * xv;
        const float zv = sZb[l * DI + d0];
        y *= zv / (1.f + __expf(-zv));
        sY[l * DI + d0] = y;  // in-place over delta (same thread, same slot)
      }
      // channel d1
      {
        const float dl = sDelta[l * DI + d1];
        const float xv = sXb[l * DI + d1];
        const float dlx = dl * xv;
        float acc = 0.f;
#pragma unroll
        for (int n = 0; n < 16; ++n) {
          h1[n] = __expf(dl * a1[n]) * h1[n] + dlx * bn[n];
          acc += h1[n] * cn[n];
        }
        float y = acc + dsk1 * xv;
        const float zv = sZb[l * DI + d1];
        y *= zv / (1.f + __expf(-zv));
        sY[l * DI + d1] = y;
      }
    }
  }
  __syncthreads();

  // 9) stage y (f16), then out_proj: (16x512)@(512x256), A hoisted (16 frags)
  for (int i = tid; i < LSEQ * DI; i += 256) sA[i] = (_Float16)sY[i];
  __syncthreads();
  {
    v16h aF[16];
#pragma unroll
    for (int ks = 0; ks < 16; ++ks) aF[ks] = loadA(sA, DI, 0, ks * 32, lane);
    for (int t = wave; t < 16; t += 8) {
      const int n0 = t * 16;
      v8f acc = {};
#pragma unroll
      for (int ks = 0; ks < 16; ++ks)
        acc = wmma_f16(aF[ks], loadB(w_outproj, DI, n0, ks * 32, lane), acc);
      const int n = n0 + (lane & 15), mh = (lane >> 4) * 8;
#pragma unroll
      for (int r = 0; r < 8; ++r) sAtt[(mh + r) * DM + n] = acc[r];
    }
  }
  __syncthreads();

  // 10) residual: x += g_m * att
  {
    const float* g_m = mrow + 512;
    for (int i = tid; i < LSEQ * DM; i += 256)
      sXres[i] += g_m[i & 255] * sAtt[i];
  }
  __syncthreads();

  // 11) LN + modulate (mlp branch)
  layernorm_mod(sXres, sA, mrow + 768, mrow + 1024, wave, lane);
  __syncthreads();

  // 12) fc1 + tanh-GELU -> sXb[16][256], A hoisted
  {
    v16h aF[8];
#pragma unroll
    for (int ks = 0; ks < 8; ++ks) aF[ks] = loadA(sA, DM, 0, ks * 32, lane);
    for (int t = wave; t < 16; t += 8) {
      const int n0 = t * 16;
      v8f acc = {};
#pragma unroll
      for (int ks = 0; ks < 8; ++ks)
        acc = wmma_f16(aF[ks], loadB(w_fc1, DM, n0, ks * 32, lane), acc);
      const int n = n0 + (lane & 15), mh = (lane >> 4) * 8;
#pragma unroll
      for (int r = 0; r < 8; ++r) {
        const float v = acc[r] + fc1_b[n];
        const float t3 = 0.7978845608028654f * (v + 0.044715f * v * v * v);
        sXb[(mh + r) * DM + n] = 0.5f * v * (1.f + tanhf(t3));
      }
    }
  }
  __syncthreads();
  for (int i = tid; i < LSEQ * DM; i += 256) sA[i] = (_Float16)sXb[i];
  __syncthreads();

  // 13) fc2 + gated residual + pixel-unshuffle scatter to output, A hoisted
  {
    const float* g_p = mrow + 1280;
    v16h aF[8];
#pragma unroll
    for (int ks = 0; ks < 8; ++ks) aF[ks] = loadA(sA, DM, 0, ks * 32, lane);
    for (int t = wave; t < 16; t += 8) {
      const int n0 = t * 16;
      v8f acc = {};
#pragma unroll
      for (int ks = 0; ks < 8; ++ks)
        acc = wmma_f16(aF[ks], loadB(w_fc2, DM, n0, ks * 32, lane), acc);
      const int n = n0 + (lane & 15), mh = (lane >> 4) * 8;
#pragma unroll
      for (int r = 0; r < 8; ++r) {
        const int m = mh + r, d = n;
        const float res = sXres[m * DM + d] + g_p[d] * (acc[r] + fc2_b[d]);
        const int c = d >> 6, r1 = (d >> 3) & 7, r2 = d & 7;
        const int hc = m >> 2, wc = m & 3;
        out[(((size_t)b * 4 + c) * 32 + hc * 8 + r1) * 32 + wc * 8 + r2] = res;
      }
    }
  }
}

// ---------------------------------------------------------------------------
extern "C" void kernel_launch(void* const* d_in, const int* in_sizes, int n_in,
                              void* d_out, int out_size, void* d_ws,
                              size_t ws_size, hipStream_t stream) {
  const float* F_clip    = (const float*)d_in[0];
  const float* F_content = (const float*)d_in[1];
  const float* fs_w      = (const float*)d_in[2];
  const float* fs_b      = (const float*)d_in[3];
  const float* in_proj_w = (const float*)d_in[4];
  const float* conv_w    = (const float*)d_in[5];
  const float* conv_b    = (const float*)d_in[6];
  const float* x_proj_w  = (const float*)d_in[7];
  const float* dt_proj_w = (const float*)d_in[8];
  const float* dt_proj_b = (const float*)d_in[9];
  const float* A_log     = (const float*)d_in[10];
  const float* D_skip    = (const float*)d_in[11];
  const float* out_proj_w= (const float*)d_in[12];
  const float* fc1_w     = (const float*)d_in[13];
  const float* fc1_b     = (const float*)d_in[14];
  const float* fc2_w     = (const float*)d_in[15];
  const float* fc2_b     = (const float*)d_in[16];
  float* out = (float*)d_out;

  // workspace layout: f16 weights then f32 mods buffer
  _Float16* f16b = (_Float16*)d_ws;
  size_t o = 0;
  _Float16* fclip16   = f16b + o; o += (size_t)512 * 512;
  _Float16* fsw16     = f16b + o; o += (size_t)1536 * 512;
  _Float16* inproj16  = f16b + o; o += (size_t)1024 * 256;
  _Float16* xproj16   = f16b + o; o += (size_t)48 * 512;
  _Float16* dtpad16   = f16b + o; o += (size_t)512 * 32;
  _Float16* outproj16 = f16b + o; o += (size_t)256 * 512;
  _Float16* fc116     = f16b + o; o += (size_t)256 * 256;
  _Float16* fc216     = f16b + o; o += (size_t)256 * 256;
  size_t f16bytes = (o * sizeof(_Float16) + 255) & ~(size_t)255;
  float* mods = (float*)((char*)d_ws + f16bytes);

  auto cvt = [&](const float* s, _Float16* d, int n) {
    cvt_f16<<<(n + 255) / 256, 256, 0, stream>>>(s, d, n);
  };
  cvt(F_clip, fclip16, 512 * 512);
  cvt(fs_w, fsw16, 1536 * 512);
  cvt(in_proj_w, inproj16, 1024 * 256);
  cvt(x_proj_w, xproj16, 48 * 512);
  cvt(out_proj_w, outproj16, 256 * 512);
  cvt(fc1_w, fc116, 256 * 256);
  cvt(fc2_w, fc216, 256 * 256);
  cvt_dtpad<<<(512 * 32 + 255) / 256, 256, 0, stream>>>(dt_proj_w, dtpad16);

  // 512x1536 modulation GEMM: 3072 tiles / 8 waves per block
  mods_gemm<<<384, 256, 0, stream>>>(fclip16, fsw16, fs_b, mods);

  // fused per-batch block
  const size_t lds = (size_t)LSEQ * DM * 4 +  // sXres
                     (size_t)LSEQ * DI * 2 +  // sA f16
                     (size_t)LSEQ * DI * 4 +  // sXb
                     (size_t)LSEQ * DI * 4 +  // sZb (alias sAtt)
                     (size_t)LSEQ * DI * 4 +  // sDelta (alias sY)
                     (size_t)LSEQ * 48 * 4;   // sDbl
  (void)hipFuncSetAttribute((const void*)fused_block,
                            hipFuncAttributeMaxDynamicSharedMemorySize,
                            (int)lds);
  fused_block<<<BATCH, 256, lds, stream>>>(
      F_content, mods, inproj16, conv_w, conv_b, xproj16, dtpad16, dt_proj_b,
      A_log, D_skip, outproj16, fc116, fc1_b, fc216, fc2_b, out);
}
